// TernaryLinear_77146202570962
// MI455X (gfx1250) — matrix operand order repaired and checked by reading, not compiled, not run on hardware
//
#include <hip/hip_runtime.h>

#define TOKENS 8192
#define IN_F   4096
#define OUT_F  4096

#define TM 128
#define TN 128
#define TK 32
#define LDT 40                 // padded LDS row stride (halves): conflict-free b128 loads
#define NKT (IN_F / TK)        // 128 k-iterations

typedef __attribute__((ext_vector_type(8)))  __bf16 v8bf;
typedef __attribute__((ext_vector_type(16))) __bf16 v16bf;
typedef __attribute__((ext_vector_type(4)))  float  v4f;
typedef __attribute__((ext_vector_type(8)))  float  v8f;
typedef __attribute__((ext_vector_type(4)))  int    v4i;

#if __has_builtin(__builtin_amdgcn_global_load_async_to_lds_b128)
#define ASYNC_LDS 1
#else
#define ASYNC_LDS 0
#endif

__device__ __forceinline__ __bf16 f32_to_bf16(float f) {
  unsigned u = __builtin_bit_cast(unsigned, f);
  unsigned r = u + 0x7FFFu + ((u >> 16) & 1u);     // round-to-nearest-even
  unsigned short h = (unsigned short)(r >> 16);
  return __builtin_bit_cast(__bf16, h);
}

// copy 16 bytes global -> LDS (async on CDNA5, sync fallback otherwise)
__device__ __forceinline__ void cp16(__bf16* lds, const __bf16* g) {
#if ASYNC_LDS
  __builtin_amdgcn_global_load_async_to_lds_b128(
      (__attribute__((address_space(1))) v4i*)g,
      (__attribute__((address_space(3))) v4i*)lds, 0, 0);
#else
  *(v8bf*)lds = *(const v8bf*)g;
#endif
}

__device__ __forceinline__ void wait_async_le4() {
#if ASYNC_LDS
#if __has_builtin(__builtin_amdgcn_s_wait_asynccnt)
  __builtin_amdgcn_s_wait_asynccnt(4);
#else
  asm volatile("s_wait_asynccnt 0x4" ::: "memory");
#endif
#endif
}

__device__ __forceinline__ void wait_async_le0() {
#if ASYNC_LDS
#if __has_builtin(__builtin_amdgcn_s_wait_asynccnt)
  __builtin_amdgcn_s_wait_asynccnt(0);
#else
  asm volatile("s_wait_asynccnt 0x0" ::: "memory");
#endif
#endif
}

// ---------------- Stage 1: deterministic sum(|W|) ----------------
__global__ __launch_bounds__(256) void ternlin_absum_partial(
    const float* __restrict__ w, float* __restrict__ partials) {
  __shared__ float sm[256];
  const int tid = threadIdx.x;
  const size_t base = (size_t)blockIdx.x * 16384 + tid;
  float acc = 0.0f;
  #pragma unroll 8
  for (int i = 0; i < 64; ++i) acc += fabsf(w[base + (size_t)i * 256]);
  sm[tid] = acc;
  __syncthreads();
  for (int off = 128; off > 0; off >>= 1) {
    if (tid < off) sm[tid] += sm[tid + off];
    __syncthreads();
  }
  if (tid == 0) partials[blockIdx.x] = sm[0];
}

__global__ __launch_bounds__(256) void ternlin_absum_final(
    const float* __restrict__ partials, float* __restrict__ delta_out) {
  __shared__ float sm[256];
  const int tid = threadIdx.x;
  float acc = 0.0f;
  for (int i = tid; i < 1024; i += 256) acc += partials[i];
  sm[tid] = acc;
  __syncthreads();
  for (int off = 128; off > 0; off >>= 1) {
    if (tid < off) sm[tid] += sm[tid + off];
    __syncthreads();
  }
  if (tid == 0) delta_out[0] = (sm[0] / 16777216.0f) * 0.7f;
}

// ---------------- Stage 2a: ternarize W -> bf16 {-1,0,+1} (8 elems/thread) ----------------
__global__ __launch_bounds__(256) void ternlin_ternarize(
    const float* __restrict__ w, const float* __restrict__ delta_p,
    __bf16* __restrict__ wq) {
  const float delta = delta_p[0];
  const size_t i = ((size_t)blockIdx.x * 256 + threadIdx.x) * 8;
  const v4f a = *(const v4f*)(w + i);
  const v4f b = *(const v4f*)(w + i + 4);
  v8bf q;
  #pragma unroll
  for (int j = 0; j < 4; ++j) {
    const float f0 = a[j];
    const float f1 = b[j];
    q[j]     = f32_to_bf16((fabsf(f0) > delta) ? (f0 > 0.0f ? 1.0f : -1.0f) : 0.0f);
    q[j + 4] = f32_to_bf16((fabsf(f1) > delta) ? (f1 > 0.0f ? 1.0f : -1.0f) : 0.0f);
  }
  *(v8bf*)(wq + i) = q;
}

// ---------------- Stage 2b: convert x -> bf16 once (8 elems/thread) ----------------
__global__ __launch_bounds__(256) void ternlin_cvt_x(
    const float* __restrict__ x, __bf16* __restrict__ xq) {
  const size_t i = ((size_t)blockIdx.x * 256 + threadIdx.x) * 8;
  const v4f a = *(const v4f*)(x + i);
  const v4f b = *(const v4f*)(x + i + 4);
  v8bf r;
  #pragma unroll
  for (int j = 0; j < 4; ++j) {
    r[j]     = f32_to_bf16(a[j]);
    r[j + 4] = f32_to_bf16(b[j]);
  }
  *(v8bf*)(xq + i) = r;
}

// ---------------- Stage 3: WMMA GEMM  y = alpha*(xq @ wq^T) + bias ----------------
// 256 threads = 8 wave32s (2 in M x 4 in N); block tile 128x128, wave tile 64x32.
// Triple-buffered global->LDS async pipeline, 2 tiles in flight,
// one s_wait_asynccnt + one barrier per K-iteration.
__global__ __launch_bounds__(256) void ternlin_gemm(
    const __bf16* __restrict__ Xq, const __bf16* __restrict__ Wq,
    const float* __restrict__ alpha_p, const float* __restrict__ bias,
    float* __restrict__ Y) {
  __shared__ __bf16 As[3][TM * LDT];
  __shared__ __bf16 Bs[3][TM * LDT];

  const int tid  = threadIdx.x;
  const int lane = tid & 31;
  const int wave = tid >> 5;
  const int wm = wave & 1;       // 0..1  (M)
  const int wn = wave >> 1;      // 0..3  (N)
  const int m_block = blockIdx.y * TM;
  const int n_block = blockIdx.x * TN;

  // tile-loader mapping: each thread owns 16 consecutive K of one row (16B x2 per matrix)
  const int lrow = tid >> 1;            // 0..127
  const int lcol = (tid & 1) * 16;      // 0 or 16

  const __bf16* Ag = Xq + (size_t)(m_block + lrow) * IN_F + lcol;
  const __bf16* Bg = Wq + (size_t)(n_block + lrow) * IN_F + lcol;
  const int lds_off = lrow * LDT + lcol;

  // prologue: issue tiles 0 and 1 (4 async b128 per thread per tile)
  #pragma unroll
  for (int t = 0; t < 2; ++t) {
    const size_t gk = (size_t)t * TK;
    cp16(&As[t][lds_off],     Ag + gk);
    cp16(&As[t][lds_off + 8], Ag + gk + 8);
    cp16(&Bs[t][lds_off],     Bg + gk);
    cp16(&Bs[t][lds_off + 8], Bg + gk + 8);
  }

  v8f acc[4][2];
  #pragma unroll
  for (int mi = 0; mi < 4; ++mi)
    #pragma unroll
    for (int ni = 0; ni < 2; ++ni) acc[mi][ni] = 0.0f;

  // fragment addressing (contiguous K order k = 16*kh + s for both A and B, wave32)
  const int kh    = lane >> 4;               // half-wave id
  const int arow  = wm * 64 + (lane & 15);   // A row within tile
  const int aoff1 = kh * 16;                 // K {0..7}/{16..23}
  const int aoff2 = kh * 16 + 8;             // K {8..15}/{24..31}
  const int bcol  = wn * 32 + (lane & 15);   // B row (output feature)
  const int boff  = kh * 16;                 // K {0..15}/{16..31}

  int cur = 0;        // kt % 3
  int nxt2 = 2;       // (kt+2) % 3
  for (int kt = 0; kt < NKT; ++kt) {
    // complete tile kt (tiles kt and kt+1 in flight -> wait <=4 leaves kt+1 pending)
    if (kt + 1 < NKT) wait_async_le4();
    else              wait_async_le0();
    __syncthreads();   // publish tile kt; also: all waves done reading buf nxt2

    if (kt + 2 < NKT) {  // issue tile kt+2 into buffer (kt+2)%3
      const size_t gk = (size_t)(kt + 2) * TK;
      cp16(&As[nxt2][lds_off],     Ag + gk);
      cp16(&As[nxt2][lds_off + 8], Ag + gk + 8);
      cp16(&Bs[nxt2][lds_off],     Bg + gk);
      cp16(&Bs[nxt2][lds_off + 8], Bg + gk + 8);
    }

    // compute on buffer cur = kt % 3
    v16bf af[4], bf[2];
    #pragma unroll
    for (int mi = 0; mi < 4; ++mi) {
      const __bf16* p = &As[cur][(arow + mi * 16) * LDT];
      v8bf lo = *(const v8bf*)(p + aoff1);
      v8bf hi = *(const v8bf*)(p + aoff2);
      af[mi] = __builtin_shufflevector(lo, hi, 0, 1, 2, 3, 4, 5, 6, 7,
                                       8, 9, 10, 11, 12, 13, 14, 15);
    }
    #pragma unroll
    for (int ni = 0; ni < 2; ++ni) {
      const __bf16* p = &Bs[cur][(bcol + ni * 16) * LDT + boff];
      v8bf lo = *(const v8bf*)(p);
      v8bf hi = *(const v8bf*)(p + 8);
      bf[ni] = __builtin_shufflevector(lo, hi, 0, 1, 2, 3, 4, 5, 6, 7,
                                       8, 9, 10, 11, 12, 13, 14, 15);
    }
    #pragma unroll
    for (int mi = 0; mi < 4; ++mi)
      #pragma unroll
      for (int ni = 0; ni < 2; ++ni)
        acc[mi][ni] = __builtin_amdgcn_wmma_f32_16x16x32_bf16(
            false, af[mi], false, bf[ni], (short)0, acc[mi][ni], false, false);

    cur  = (cur  == 2) ? 0 : cur + 1;
    nxt2 = (nxt2 == 2) ? 0 : nxt2 + 1;
  }

  // epilogue: y = alpha*acc + bias   (C/D layout: VGPR r -> M = r + kh*8, lane%16 -> N)
  const float alpha = alpha_p[0];
  #pragma unroll
  for (int ni = 0; ni < 2; ++ni) {
    const int col = n_block + wn * 32 + ni * 16 + (lane & 15);
    const float bv = bias[col];
    #pragma unroll
    for (int mi = 0; mi < 4; ++mi) {
      const int row0 = m_block + wm * 64 + mi * 16 + kh * 8;
      #pragma unroll
      for (int r = 0; r < 8; ++r)
        Y[(size_t)(row0 + r) * OUT_F + col] = alpha * acc[mi][ni][r] + bv;
    }
  }
}

extern "C" void kernel_launch(void* const* d_in, const int* in_sizes, int n_in,
                              void* d_out, int out_size, void* d_ws, size_t ws_size,
                              hipStream_t stream) {
  (void)in_sizes; (void)n_in; (void)out_size; (void)ws_size;
  const float* x      = (const float*)d_in[0];
  const float* weight = (const float*)d_in[1];
  const float* alpha  = (const float*)d_in[2];
  const float* bias   = (const float*)d_in[3];
  float* out = (float*)d_out;

  // workspace layout: [wq bf16: 32 MB][xq bf16: 64 MB][red: delta + 1024 partials]
  char* ws = (char*)d_ws;
  __bf16* wq  = (__bf16*)ws;
  __bf16* xq  = (__bf16*)(ws + (size_t)OUT_F * IN_F * sizeof(__bf16));
  float*  red = (float*)(ws + (size_t)OUT_F * IN_F * sizeof(__bf16)
                            + (size_t)TOKENS * IN_F * sizeof(__bf16));
  // red[0] = delta, red[1..1024] = partial sums

  ternlin_absum_partial<<<1024, 256, 0, stream>>>(weight, red + 1);
  ternlin_absum_final<<<1, 256, 0, stream>>>(red + 1, red);
  ternlin_ternarize<<<(OUT_F * IN_F) / 2048, 256, 0, stream>>>(weight, red, wq);
  ternlin_cvt_x<<<(TOKENS * IN_F) / 2048, 256, 0, stream>>>(x, xq);

  dim3 grid(OUT_F / TN, TOKENS / TM);
  ternlin_gemm<<<grid, 256, 0, stream>>>(xq, wq, alpha, bias, out);
}
